// MultiQueryAttention_9131100471593
// MI455X (gfx1250) — matrix operand order repaired and checked by reading, not compiled, not run on hardware
//
#include <hip/hip_runtime.h>

// ---------- types ----------
typedef __attribute__((ext_vector_type(16))) __bf16 v16bf;
typedef __attribute__((ext_vector_type(8)))  float  v8f;

__device__ __forceinline__ unsigned short f2bf(float f) {
    unsigned int u = __float_as_uint(f);
    u += 0x7fffu + ((u >> 16) & 1u);      // round-to-nearest-even
    return (unsigned short)(u >> 16);
}
__device__ __forceinline__ __bf16 bf_bits(unsigned short b) {
    union { unsigned short u; __bf16 h; } c; c.u = b; return c.h;
}
// Low 32 bits of a generic pointer into the LDS aperture = LDS byte address
// (see CDNA5 ISA 10.2).
__device__ __forceinline__ unsigned lds_addr32(const void* p) {
    return (unsigned)(uintptr_t)p;
}
// gfx1250 async copy: global -> LDS, 16B per lane, ASYNCcnt-tracked, no VGPR data.
__device__ __forceinline__ void async_ld_b128(unsigned lds, unsigned goff_bytes,
                                              const void* sbase) {
    asm volatile("global_load_async_to_lds_b128 %0, %1, %2"
                 :: "v"(lds), "v"(goff_bytes), "s"(sbase) : "memory");
}
__device__ __forceinline__ void wait_async0() {
    asm volatile("s_wait_asynccnt 0" ::: "memory");
}

// ---------- DPP cross-lane (VALU pipe, no LDS traffic) ----------
constexpr unsigned dpp8_sel_xor(unsigned x) {   // sel_i = i ^ x, packed 3b each
    unsigned s = 0;
    for (unsigned i = 0; i < 8; ++i) s |= ((i ^ x) & 7u) << (3 * i);
    return s;
}
template <unsigned X>
__device__ __forceinline__ float dpp_xor_small(float v) {   // X in {1,2,4}
    return __int_as_float(
        __builtin_amdgcn_mov_dpp8(__float_as_int(v), dpp8_sel_xor(X)));
}
__device__ __forceinline__ float dpp_xor8(float v) {        // ROW_ROR:8 == xor 8
    return __int_as_float(__builtin_amdgcn_update_dpp(
        0, __float_as_int(v), 0x128 /*ROW_ROR:8*/, 0xF, 0xF, true));
}
__device__ __forceinline__ float red16_max(float v) {
    v = fmaxf(v, dpp_xor_small<1>(v));
    v = fmaxf(v, dpp_xor_small<2>(v));
    v = fmaxf(v, dpp_xor_small<4>(v));
    v = fmaxf(v, dpp_xor8(v));
    return v;
}
__device__ __forceinline__ float red16_sum(float v) {
    v += dpp_xor_small<1>(v);
    v += dpp_xor_small<2>(v);
    v += dpp_xor_small<4>(v);
    v += dpp_xor8(v);
    return v;
}

// ---------- 0a) fp32 -> bf16 (straight) ----------
__global__ __launch_bounds__(256) void cvt_f32_bf16(const float* __restrict__ in,
                                                    unsigned short* __restrict__ out,
                                                    int n) {
    int i = (blockIdx.x * blockDim.x + threadIdx.x) * 4;
    if (i + 3 < n) {
        float4 v = *reinterpret_cast<const float4*>(in + i);
        out[i + 0] = f2bf(v.x);
        out[i + 1] = f2bf(v.y);
        out[i + 2] = f2bf(v.z);
        out[i + 3] = f2bf(v.w);
    }
}

// ---------- 0b) fp32 [R x C] -> bf16 transposed [C x R] (weight B-operands) ----------
__global__ __launch_bounds__(256) void cvt_f32_bf16_T(const float* __restrict__ in,
                                                      unsigned short* __restrict__ out,
                                                      int R, int C) {
    int i = blockIdx.x * blockDim.x + threadIdx.x;
    if (i < R * C) {
        int r = i / C, c = i % C;
        out[(size_t)c * R + r] = f2bf(in[i]);
    }
}

// ---------- 1) tiled bf16 WMMA GEMM: C[M,N] = A[M,K] * Bt[N,K]^T ----------
// A row-major [M x K]; Bt row-major [N x K] (pre-transposed weights).
// Block = 128 threads (4 waves) -> 128x64 tile; wave w owns rows [w*32, w*32+32).
// Double-buffered GLOBAL_LOAD_ASYNC_TO_LDS_B128 pipeline; fragments contiguous/lane.
template <bool F32OUT, bool STORE_T>
__global__ __launch_bounds__(128) void gemm_bf16(const unsigned short* __restrict__ A,
                                                 const unsigned short* __restrict__ Bt,
                                                 void* __restrict__ Cp,
                                                 int M, int N, int K) {
    constexpr int PSTR = 48;                 // panel row stride (96B, 16B-aligned)
    constexpr int ASZ  = 128 * PSTR;
    constexpr int BSZ  = 64 * PSTR;
    __shared__ alignas(16) unsigned short lA[2 * ASZ];
    __shared__ alignas(16) unsigned short lB[2 * BSZ];

    const int tid = threadIdx.x;
    const int w   = tid >> 5;
    const int mn  = tid & 15;
    const int hi  = (tid >> 4) & 1;
    const int m0  = blockIdx.x * 128;
    const int n0  = blockIdx.y * 64;

    auto issue_panel = [&](int k0, int buf) {
        unsigned short* pA = lA + buf * ASZ;
        unsigned short* pB = lB + buf * BSZ;
#pragma unroll
        for (int c = 0; c < 4; ++c) {               // A: 128x32 = 512 b128 chunks
            int chunk = tid * 4 + c;
            int ar = chunk >> 2, ac = (chunk & 3) * 8;
            async_ld_b128(lds_addr32(&pA[ar * PSTR + ac]),
                          (unsigned)((m0 + ar) * K + k0 + ac) * 2u, A);
        }
#pragma unroll
        for (int c = 0; c < 2; ++c) {               // Bt: 64x32 = 256 b128 chunks
            int chunk = tid * 2 + c;
            int br = chunk >> 2, bc = (chunk & 3) * 8;
            async_ld_b128(lds_addr32(&pB[br * PSTR + bc]),
                          (unsigned)((n0 + br) * K + k0 + bc) * 2u, Bt);
        }
    };

    v8f acc[2][4];
#pragma unroll
    for (int u = 0; u < 2; ++u)
#pragma unroll
        for (int t = 0; t < 4; ++t)
            acc[u][t] = (v8f){0.f,0.f,0.f,0.f,0.f,0.f,0.f,0.f};

    issue_panel(0, 0);
    wait_async0();
    __syncthreads();

    int cur = 0;
    for (int k0 = 0; k0 < K; k0 += 32) {
        int nxt = cur ^ 1;
        if (k0 + 32 < K) issue_panel(k0 + 32, nxt);   // prefetch next panel

        const unsigned short* pA = lA + cur * ASZ;
        const unsigned short* pB = lB + cur * BSZ;
        v16bf a[2];
#pragma unroll
        for (int u = 0; u < 2; ++u)
#pragma unroll
            for (int e = 0; e < 16; ++e) {
                int k = hi * 16 + e;
                a[u][e] = bf_bits(pA[(w * 32 + u * 16 + mn) * PSTR + k]);
            }
#pragma unroll
        for (int t = 0; t < 4; ++t) {
            v16bf b;   // B[k][n] = Bt[n][k]: row t*16+mn, 16 contiguous elements
#pragma unroll
            for (int e = 0; e < 16; ++e)
                b[e] = bf_bits(pB[(t * 16 + mn) * PSTR + hi * 16 + e]);
#pragma unroll
            for (int u = 0; u < 2; ++u)
                acc[u][t] = __builtin_amdgcn_wmma_f32_16x16x32_bf16(
                                false, a[u], false, b, (short)0, acc[u][t], false, false);
        }
        wait_async0();      // next panel landed (latency hidden behind compute)
        __syncthreads();
        cur = nxt;
    }

#pragma unroll
    for (int u = 0; u < 2; ++u)
#pragma unroll
        for (int t = 0; t < 4; ++t)
#pragma unroll
            for (int r = 0; r < 8; ++r) {
                int row = m0 + w * 32 + u * 16 + hi * 8 + r;
                int col = n0 + t * 16 + mn;
                if (F32OUT)
                    reinterpret_cast<float*>(Cp)[(size_t)row * N + col] = acc[u][t][r];
                else if (STORE_T)   // store C^T: [N x M] (used for V -> Vt)
                    reinterpret_cast<unsigned short*>(Cp)[(size_t)col * M + row] =
                        f2bf(acc[u][t][r]);
                else
                    reinterpret_cast<unsigned short*>(Cp)[(size_t)row * N + col] =
                        f2bf(acc[u][t][r]);
            }
}

// ---------- 2) causal flash attention (MQA: one shared KV head) ----------
// Qg [4096 x 1024] bf16, Kg [4096 x 64] bf16, Vt [64 x 4096] bf16 (d-major).
// grid.x = B*H (b = bx/16, h = bx&15), grid.y = S/64. 4 waves, 16 q-rows each.
// Double-buffered async K/V staging; mask only on diagonal tile; DPP reductions.
__global__ __launch_bounds__(128) void mqa_attn(const unsigned short* __restrict__ Qg,
                                                const unsigned short* __restrict__ Kg,
                                                const unsigned short* __restrict__ Vt,
                                                unsigned short* __restrict__ AO) {
    constexpr int S = 2048, DM = 1024, DK = 64, STR = 72;
    constexpr int TSZ = 64 * STR;
    __shared__ alignas(16) unsigned short lK[2 * TSZ];   // [kv][d]
    __shared__ alignas(16) unsigned short lV[2 * TSZ];   // [d][kv] (from Vt)
    __shared__ alignas(16) unsigned short lP[4 * 16 * STR];

    const int tid  = threadIdx.x;
    const int w    = tid >> 5;
    const int mn   = tid & 15;
    const int hi   = (tid >> 4) & 1;
    const int b    = blockIdx.x >> 4;
    const int h    = blockIdx.x & 15;
    const int qt   = blockIdx.y;
    const int q0   = qt * 64;
    const int qrow = q0 + w * 16;

    auto issue_tiles = [&](int jt, int buf) {
#pragma unroll
        for (int c = 0; c < 4; ++c) {               // 512 b128 chunks per tile
            int chunk = tid * 4 + c;
            int r  = chunk >> 3;
            int c8 = (chunk & 7) * 8;
            async_ld_b128(lds_addr32(&lK[buf * TSZ + r * STR + c8]),
                          (unsigned)((b * S + jt * 64 + r) * DK + c8) * 2u, Kg);
            // Vt layout: [d][b*S + s] -> row r is d, cols are kv positions
            async_ld_b128(lds_addr32(&lV[buf * TSZ + r * STR + c8]),
                          (unsigned)(r * (2 * S) + b * S + jt * 64 + c8) * 2u, Vt);
        }
    };

    const float alpha = 0.125f * 1.44269504088896340736f;  // (1/sqrt(dk))*log2(e)

    // Q strip [16 x 64] as two 16x32 A-fragments; fold alpha in (re-round via f32)
    v16bf qf[2];
#pragma unroll
    for (int ks = 0; ks < 2; ++ks)
#pragma unroll
        for (int e = 0; e < 16; ++e) {
            int d = ks * 32 + hi * 16 + e;
            unsigned short qb = Qg[(size_t)(b * S + qrow + mn) * DM + h * DK + d];
            float qv = __uint_as_float((unsigned)qb << 16) * alpha;
            qf[ks][e] = bf_bits(f2bf(qv));
        }

    v8f acc[4];
    float m_run[8], l_run[8];
#pragma unroll
    for (int t = 0; t < 4; ++t) acc[t] = (v8f){0.f,0.f,0.f,0.f,0.f,0.f,0.f,0.f};
#pragma unroll
    for (int r = 0; r < 8; ++r) { m_run[r] = -3.0e38f; l_run[r] = 0.f; }

    issue_tiles(0, 0);
    wait_async0();
    __syncthreads();

    int cur = 0;
    for (int jt = 0; jt <= qt; ++jt) {
        int nxt = cur ^ 1;
        if (jt < qt) issue_tiles(jt + 1, nxt);       // prefetch next KV tile

        const unsigned short* pK = lK + cur * TSZ;
        const unsigned short* pV = lV + cur * TSZ;

        // ---- S = Q @ K^T (already scaled; 4 score tiles of 16x16) ----
        v8f sc[4];
#pragma unroll
        for (int t = 0; t < 4; ++t) {
            v8f s = (v8f){0.f,0.f,0.f,0.f,0.f,0.f,0.f,0.f};
#pragma unroll
            for (int ks = 0; ks < 2; ++ks) {
                v16bf bk;   // B[k=d][n=kv] = K[kv][d]: contiguous per lane
#pragma unroll
                for (int e = 0; e < 16; ++e)
                    bk[e] = bf_bits(pK[(t * 16 + mn) * STR + ks * 32 + hi * 16 + e]);
                s = __builtin_amdgcn_wmma_f32_16x16x32_bf16(
                        false, qf[ks], false, bk, (short)0, s, false, false);
            }
            sc[t] = s;
        }
        // ---- causal mask: only the diagonal tile needs it (uniform branch) ----
        if (jt == qt) {
#pragma unroll
            for (int t = 0; t < 4; ++t)
#pragma unroll
                for (int r = 0; r < 8; ++r) {
                    int i = qrow + hi * 8 + r;
                    int j = jt * 64 + t * 16 + mn;
                    sc[t][r] = (j <= i) ? sc[t][r] : -3.0e38f;
                }
        }
        // ---- online softmax; row reductions via DPP (VALU pipe, no LDS) ----
        float fac[8];
#pragma unroll
        for (int r = 0; r < 8; ++r) {
            float mx = red16_max(
                fmaxf(fmaxf(sc[0][r], sc[1][r]), fmaxf(sc[2][r], sc[3][r])));
            float mnew = fmaxf(m_run[r], mx);
            fac[r]   = exp2f(m_run[r] - mnew);
            m_run[r] = mnew;
            float rs = 0.f;
#pragma unroll
            for (int t = 0; t < 4; ++t) {
                float p = exp2f(sc[t][r] - mnew);
                sc[t][r] = p;
                rs += p;
            }
            l_run[r] = l_run[r] * fac[r] + red16_sum(rs);
        }
        // ---- rescale accumulators, spill P (C-layout) to wave-private LDS ----
#pragma unroll
        for (int t = 0; t < 4; ++t)
#pragma unroll
            for (int r = 0; r < 8; ++r) {
                acc[t][r] *= fac[r];
                lP[(w * 16 + hi * 8 + r) * STR + t * 16 + mn] = f2bf(sc[t][r]);
            }
        // ---- O += P @ V : reload P in A-layout (same-wave LDS is in-order) ----
        v16bf pa[2];
#pragma unroll
        for (int ks = 0; ks < 2; ++ks)
#pragma unroll
            for (int e = 0; e < 16; ++e) {
                int kv = ks * 32 + hi * 16 + e;
                pa[ks][e] = bf_bits(lP[(w * 16 + mn) * STR + kv]);
            }
#pragma unroll
        for (int t = 0; t < 4; ++t)
#pragma unroll
            for (int ks = 0; ks < 2; ++ks) {
                v16bf bv;   // B[k=kv][n=d] = lV[d][kv]: contiguous per lane
#pragma unroll
                for (int e = 0; e < 16; ++e)
                    bv[e] = bf_bits(pV[(t * 16 + mn) * STR + ks * 32 + hi * 16 + e]);
                acc[t] = __builtin_amdgcn_wmma_f32_16x16x32_bf16(
                             false, pa[ks], false, bv, (short)0, acc[t], false, false);
            }

        wait_async0();        // prefetched tiles landed (hidden behind compute)
        __syncthreads();
        cur = nxt;
    }

    // ---- epilogue: normalize, store bf16 attention output [4096 x 1024] ----
#pragma unroll
    for (int r = 0; r < 8; ++r) l_run[r] = 1.0f / l_run[r];
#pragma unroll
    for (int t = 0; t < 4; ++t)
#pragma unroll
        for (int r = 0; r < 8; ++r) {
            int srow = qrow + hi * 8 + r;
            AO[(size_t)(b * S + srow) * DM + h * DK + t * 16 + mn] =
                f2bf(acc[t][r] * l_run[r]);
        }
}

// ---------- host-side orchestration ----------
extern "C" void kernel_launch(void* const* d_in, const int* in_sizes, int n_in,
                              void* d_out, int out_size, void* d_ws, size_t ws_size,
                              hipStream_t stream) {
    (void)in_sizes; (void)n_in; (void)out_size; (void)ws_size;
    const float* x  = (const float*)d_in[0];
    // d_in[1] = attention_mask (exact causal tril; computed analytically in-kernel)
    const float* Wq = (const float*)d_in[2];
    const float* Wk = (const float*)d_in[3];
    const float* Wv = (const float*)d_in[4];
    const float* Wo = (const float*)d_in[5];
    float* out = (float*)d_out;

    constexpr int Bv = 2, Sv = 2048, DM = 1024, DK = 64;
    constexpr int M = Bv * Sv;  // 4096

    char* ws = (char*)d_ws;
    size_t off = 0;
    auto take = [&](size_t bytes) -> unsigned short* {
        char* p = ws + off;
        off += (bytes + 255) & ~(size_t)255;
        return (unsigned short*)p;
    };
    unsigned short* Xb   = take((size_t)M * DM * 2);   // x, bf16 row-major
    unsigned short* Wqt  = take((size_t)DM * DM * 2);  // weights transposed [N x K]
    unsigned short* Wkt  = take((size_t)DK * DM * 2);
    unsigned short* Wvt  = take((size_t)DK * DM * 2);
    unsigned short* Wot  = take((size_t)DM * DM * 2);
    unsigned short* Qb   = take((size_t)M * DM * 2);   // Q [4096 x 1024]
    unsigned short* Kb   = take((size_t)M * DK * 2);   // K [4096 x 64]
    unsigned short* Vtb  = take((size_t)M * DK * 2);   // V^T [64 x 4096]
    unsigned short* AOb  = take((size_t)M * DM * 2);   // attention out, bf16

    cvt_f32_bf16<<<dim3((M * DM) / 1024), dim3(256), 0, stream>>>(x, Xb, M * DM);
    cvt_f32_bf16_T<<<dim3((DM * DM) / 256), dim3(256), 0, stream>>>(Wq, Wqt, DM, DM);
    cvt_f32_bf16_T<<<dim3((DM * DK) / 256), dim3(256), 0, stream>>>(Wk, Wkt, DM, DK);
    cvt_f32_bf16_T<<<dim3((DM * DK) / 256), dim3(256), 0, stream>>>(Wv, Wvt, DM, DK);
    cvt_f32_bf16_T<<<dim3((DM * DM) / 256), dim3(256), 0, stream>>>(Wo, Wot, DM, DM);

    // Projections
    gemm_bf16<false, false><<<dim3(M / 128, DM / 64), dim3(128), 0, stream>>>(
        Xb, Wqt, Qb, M, DM, DM);
    gemm_bf16<false, false><<<dim3(M / 128, 1), dim3(128), 0, stream>>>(
        Xb, Wkt, Kb, M, DK, DM);
    gemm_bf16<false, true><<<dim3(M / 128, 1), dim3(128), 0, stream>>>(
        Xb, Wvt, Vtb, M, DK, DM);   // stored transposed: Vt [64 x 4096]

    // Flash attention (causal, shared KV head)
    mqa_attn<<<dim3(Bv * 16, Sv / 64), dim3(128), 0, stream>>>(Qb, Kb, Vtb, AOb);

    // Output projection (fp32 straight to d_out)
    gemm_bf16<true, false><<<dim3(M / 128, DM / 64), dim3(128), 0, stream>>>(
        AOb, Wot, out, M, DM, DM);
}
